// PatchEmbedPerChannel_72155450573180
// MI455X (gfx1250) — matrix operand order repaired and checked by reading, not compiled, not run on hardware
//
#include <hip/hip_runtime.h>
#include <cstdint>

typedef __attribute__((ext_vector_type(16))) _Float16 v16h;
typedef __attribute__((ext_vector_type(8)))  _Float16 v8h;
typedef __attribute__((ext_vector_type(8)))  float    v8f;

#define PS    16
#define DIM   768
#define CIN   8
#define HID   128
#define WSIZE (DIM * PS * PS)     // 196608
#define NCOL  (WSIZE + DIM)       // 197376
#define BATCH 32
#define HP    14
#define MPC   (BATCH * HP * HP)   // 6272 rows (M) per channel
#define LDA   264                 // padded LDS row stride in halves

// ---------------- Stage 1: h = relu(ce @ fc1_w + fc1_b), (8,768)x(768,128) ----------------
__global__ __launch_bounds__(1024) void fc1_kernel(const float* __restrict__ ce,
                                                   const float* __restrict__ w,
                                                   const float* __restrict__ b,
                                                   float* __restrict__ hOut) {
    int t = threadIdx.x;          // 1024 threads: c = t/128, j = t%128
    int c = t >> 7, j = t & 127;
    float acc = b[j];
    const float* cr = ce + c * DIM;
    for (int k = 0; k < DIM; ++k) acc = fmaf(cr[k], w[k * HID + j], acc);
    hOut[t] = fmaxf(acc, 0.0f);
}

// ---------------- Stage 2: y = tanh(h @ fc2_w + fc2_b); pack wts into WMMA B-fragment
// layout (f16) and bias into f32. B-fragment for (c, ntile, kchunk): 32 lanes x 16 halves,
// lane L = 16*g + n holds column d = ntile*16+n, K = kchunk*32 + 16*g + i  (i = 0..15).
__global__ __launch_bounds__(256) void fc2_pack_kernel(const float* __restrict__ h,
                                                       const float* __restrict__ w2,
                                                       const float* __restrict__ b2,
                                                       _Float16* __restrict__ wsB,
                                                       float* __restrict__ biasOut) {
    __shared__ float sh[CIN * HID];
    for (int i = threadIdx.x; i < CIN * HID; i += 256) sh[i] = h[i];
    __syncthreads();

    int j = blockIdx.x * 256 + threadIdx.x;
    if (j >= NCOL) return;

    float base = b2[j];
    float acc[CIN];
#pragma unroll
    for (int c = 0; c < CIN; ++c) acc[c] = base;
    for (int k = 0; k < HID; ++k) {
        float wv = w2[(size_t)k * NCOL + j];
#pragma unroll
        for (int c = 0; c < CIN; ++c) acc[c] = fmaf(sh[c * HID + k], wv, acc[c]);
    }
#pragma unroll
    for (int c = 0; c < CIN; ++c) {
        float yv = tanhf(acc[c]);
        if (j < WSIZE) {
            int d  = j >> 8;            // output column (0..767)
            int k2 = j & 255;           // K index  = p*16+q (0..255)
            int kc = k2 >> 5;           // K chunk of 32
            int kk = k2 & 31;
            int g  = kk >> 4;           // lane group
            int i2 = kk & 15;           // half index within lane
            int tt = d >> 4;            // 16-col N tile
            int n  = d & 15;
            int L  = (g << 4) | n;
            size_t off = (((size_t)(c * 48 + tt)) * 8 + kc) * 512 + (size_t)L * 16 + i2;
            wsB[off] = (_Float16)yv;
        } else {
            biasOut[c * DIM + (j - WSIZE)] = yv;
        }
    }
}

// ---------------- Stage 3: per-channel GEMM, M=6272, K=256, N=768, f16 WMMA, f32 accum.
// Block = 8 waves, tile 64(M) x 128(N); full K staged in LDS as f16.
__global__ __launch_bounds__(256) void patch_wmma_kernel(const float* __restrict__ x,
                                                         const _Float16* __restrict__ wsB,
                                                         const float* __restrict__ biasv,
                                                         float* __restrict__ out) {
    const int c     = blockIdx.z;
    const int Mbase = blockIdx.y * 64;
    const int ntile = blockIdx.x * 8;            // first 16-col N tile of this block

    __shared__ _Float16 ldsA[64 * LDA];
    const int t = threadIdx.x;

    // ---- Load A tile: 64 rows x 256 K (fp32 patches -> f16 in LDS) ----
    {
        int rloc = t >> 2;                       // 0..63 : row within tile
        int k0   = (t & 3) << 6;                 // 0/64/128/192
        int r    = Mbase + rloc;                 // global row within channel
        int b    = r / 196;
        int hw   = r - b * 196;
        int hh   = hw / 14;
        int ww   = hw - hh * 14;
        const float* xrow = x + (((size_t)(b * CIN + c) * 224 + hh * 16) * 224 + ww * 16);
#pragma unroll
        for (int s = 0; s < 4; ++s) {
            int k = k0 + (s << 4);               // K = p*16 + q, 16 consecutive q
            int p = k >> 4;
            const float4* src = (const float4*)(xrow + p * 224);
            float4 f0 = src[0], f1 = src[1], f2 = src[2], f3 = src[3];
            v8h h0, h1;
            h0[0]=(_Float16)f0.x; h0[1]=(_Float16)f0.y; h0[2]=(_Float16)f0.z; h0[3]=(_Float16)f0.w;
            h0[4]=(_Float16)f1.x; h0[5]=(_Float16)f1.y; h0[6]=(_Float16)f1.z; h0[7]=(_Float16)f1.w;
            h1[0]=(_Float16)f2.x; h1[1]=(_Float16)f2.y; h1[2]=(_Float16)f2.z; h1[3]=(_Float16)f2.w;
            h1[4]=(_Float16)f3.x; h1[5]=(_Float16)f3.y; h1[6]=(_Float16)f3.z; h1[7]=(_Float16)f3.w;
            *(v8h*)&ldsA[rloc * LDA + k]     = h0;
            *(v8h*)&ldsA[rloc * LDA + k + 8] = h1;
        }
    }
    __syncthreads();

    // ---- WMMA: each wave does a 32x32 tile (2x2 of 16x16), 8 K-steps ----
    const int wave = t >> 5, lane = t & 31;
    const int wm = wave >> 2, wn = wave & 3;     // 2 (M) x 4 (N) waves
    const int g  = lane >> 4, l16 = lane & 15;

    v8f acc[2][2] = {};
    const int nt0 = ntile + wn * 2;              // wave's first 16-col tile
    const _Float16* bfragBase = wsB + ((size_t)(c * 48 + nt0)) * 8 * 512 + (size_t)lane * 16;

#pragma unroll
    for (int kc = 0; kc < 8; ++kc) {
        // A fragments: half i -> K = kc*32+16g+i (matches packed B)
        int kA   = kc * 32 + (g << 4);
        int row0 = (wm << 5) + l16;
        v8h a0lo = *(const v8h*)&ldsA[row0 * LDA + kA];
        v8h a0hi = *(const v8h*)&ldsA[row0 * LDA + kA + 8];
        v8h a1lo = *(const v8h*)&ldsA[(row0 + 16) * LDA + kA];
        v8h a1hi = *(const v8h*)&ldsA[(row0 + 16) * LDA + kA + 8];
        v16h a0 = __builtin_shufflevector(a0lo, a0hi, 0,1,2,3,4,5,6,7,8,9,10,11,12,13,14,15);
        v16h a1 = __builtin_shufflevector(a1lo, a1hi, 0,1,2,3,4,5,6,7,8,9,10,11,12,13,14,15);

        v16h b0 = *(const v16h*)(bfragBase + (size_t)kc * 512);
        v16h b1 = *(const v16h*)(bfragBase + 8 * 512 + (size_t)kc * 512);

        acc[0][0] = __builtin_amdgcn_wmma_f32_16x16x32_f16(false, a0, false, b0, (short)0, acc[0][0], false, false);
        acc[0][1] = __builtin_amdgcn_wmma_f32_16x16x32_f16(false, a0, false, b1, (short)0, acc[0][1], false, false);
        acc[1][0] = __builtin_amdgcn_wmma_f32_16x16x32_f16(false, a1, false, b0, (short)0, acc[1][0], false, false);
        acc[1][1] = __builtin_amdgcn_wmma_f32_16x16x32_f16(false, a1, false, b1, (short)0, acc[1][1], false, false);
    }

    // ---- Epilogue: bias add + scattered store to (b, c*196+hw, d) ----
#pragma unroll
    for (int ni = 0; ni < 2; ++ni) {
        int d = (nt0 + ni) * 16 + l16;
        float bv = biasv[c * DIM + d];
#pragma unroll
        for (int mi = 0; mi < 2; ++mi) {
#pragma unroll
            for (int vr = 0; vr < 8; ++vr) {
                int m  = vr + (g << 3);          // C/D layout: M = vgpr + 8*(lane/16)
                int rg = Mbase + (wm << 5) + (mi << 4) + m;
                int b  = rg / 196;
                int hw = rg - b * 196;
                out[(size_t)b * (1568 * 768) + (size_t)(c * 196 + hw) * 768 + d] =
                    acc[mi][ni][vr] + bv;
            }
        }
    }
}

// ---------------- Host launcher ----------------
extern "C" void kernel_launch(void* const* d_in, const int* in_sizes, int n_in,
                              void* d_out, int out_size, void* d_ws, size_t ws_size,
                              hipStream_t stream) {
    const float* x     = (const float*)d_in[0];
    // d_in[1] = channels (unused by the reference computation)
    const float* ce    = (const float*)d_in[2];
    const float* fc1_w = (const float*)d_in[3];
    const float* fc1_b = (const float*)d_in[4];
    const float* fc2_w = (const float*)d_in[5];
    const float* fc2_b = (const float*)d_in[6];
    float* out = (float*)d_out;

    // Workspace layout: h (1024 f32) | bias (8*768 f32) | packed B frags (8*768*256 f16 = 3 MB)
    float*    h     = (float*)d_ws;
    float*    biasv = h + CIN * HID;
    _Float16* wsB   = (_Float16*)(biasv + CIN * DIM);

    fc1_kernel<<<1, 1024, 0, stream>>>(ce, fc1_w, fc1_b, h);
    fc2_pack_kernel<<<(NCOL + 255) / 256, 256, 0, stream>>>(h, fc2_w, fc2_b, wsB, biasv);

    dim3 grid(DIM / 128, MPC / 64, CIN);         // (6, 98, 8)
    patch_wmma_kernel<<<grid, 256, 0, stream>>>(x, wsB, biasv, out);
}
